// AttentionHead_50165218018181
// MI455X (gfx1250) — matrix operand order repaired and checked by reading, not compiled, not run on hardware
//
#include <hip/hip_runtime.h>
#include <hip/hip_bf16.h>

// ---------------------------------------------------------------------------
// Single-head causal attention for MI455X (gfx1250, wave32, WMMA).
//   x:[4,2048,1024] f32, Wq/Wk/Wv:[1024,64] f32  ->  out:[4,2048,64] f32
// All matmuls run on v_wmma_f32_16x16x32_f16 (f16 in, f32 accumulate).
// V tiles for the PV step are prefetched with global_load_async_to_lds_b128
// (ASYNCcnt) one chunk ahead, double-buffered in LDS.
// ---------------------------------------------------------------------------

typedef __attribute__((ext_vector_type(16))) _Float16 v16h;
typedef __attribute__((ext_vector_type(8)))  float    v8f;

#define B_     4
#define T_     2048
#define C_     1024
#define H_     64
#define QSCALE 0.03125f           // 1/sqrt(1024)

// A-fragment (16x32 f16) K index for vgpr-slot j (0..7), lane group kg (0/1).
__device__ __forceinline__ int a_kidx(int j, int kg) {
    return kg * 16 + 2 * j;
}

// Issue async copy of one 32-byte-per-lane B fragment: global -> LDS.
// IOFFSET is added to BOTH the global and LDS addresses (ISA 10.7 pseudocode),
// so a single offset:16 covers the second half of the fragment.
__device__ __forceinline__ void vfrag_async(const v16h* gsrc, _Float16* ldst) {
    unsigned long long ga = (unsigned long long)(uintptr_t)gsrc;
    unsigned int       la = (unsigned int)(uintptr_t)ldst;   // low 32 = LDS off
    asm volatile("global_load_async_to_lds_b128 %0, %1, off\n\t"
                 "global_load_async_to_lds_b128 %0, %1, off offset:16"
                 :: "v"(la), "v"(ga) : "memory");
}

// ---------------------------------------------------------------------------
// Kernel 1: swizzle Wq/Wk/Wv (f32) into WMMA B-fragment layout (f16).
// Fragment index = ((kc*12 + ct)*32 + lane)  (kc-major: the 12 fragments per
// K step are contiguous for the GEMM's streaming access).
//   ct 0..3 -> Wq cols, 4..7 -> Wk, 8..11 -> Wv ; kc = K chunk of 32.
// B-frag layout: lane n = lane&15 (N), K = (lane>>4)*16 + 2j + e.
// ---------------------------------------------------------------------------
__global__ void __launch_bounds__(256)
swz_w(const float* __restrict__ Wq, const float* __restrict__ Wk,
      const float* __restrict__ Wv, _Float16* __restrict__ Wswz) {
    int tid = blockIdx.x * blockDim.x + threadIdx.x;
    if (tid >= 12 * 32 * 32) return;
    int lane = tid & 31;
    int kc   = (tid >> 5) & 31;
    int ct   = tid >> 10;
    const float* W = (ct < 4) ? Wq : (ct < 8 ? Wk : Wv);
    int n  = lane & 15, kg = lane >> 4;
    int col = (ct & 3) * 16 + n;
    v16h o;
#pragma unroll
    for (int j = 0; j < 8; ++j) {
        int k = kc * 32 + kg * 16 + 2 * j;
        o[2 * j]     = (_Float16)W[(size_t)k * H_ + col];
        o[2 * j + 1] = (_Float16)W[(size_t)(k + 1) * H_ + col];
    }
    ((v16h*)Wswz)[(size_t)(kc * 12 + ct) * 32 + lane] = o;
}

// ---------------------------------------------------------------------------
// Kernel 2: QKV projection. One wave per 16-ROW tile computing all 12
// 16-wide N tiles (q0..3,k0..3,v0..3): x is read exactly once, and each
// A-fragment load feeds 12 WMMAs. All 12 B fragments are materialized as
// distinct values BEFORE the WMMA chain; __launch_bounds__(256,1) relaxes
// the occupancy target so the allocator can keep them all live (~220 VGPRs)
// and the scheduler can clause the loads and stagger the waits across the
// WMMA chain instead of load->wait(0)->wmma serialization.
// Q is pre-scaled by 1/sqrt(C) at store. 512 waves, 8 waves/block.
// ---------------------------------------------------------------------------
__global__ void __launch_bounds__(256, 1)
qkv_proj(const float* __restrict__ x, const _Float16* __restrict__ Wswz,
         _Float16* __restrict__ Q, _Float16* __restrict__ K,
         _Float16* __restrict__ V) {
    int lane = threadIdx.x & 31;
    int wave = (blockIdx.x * blockDim.x + threadIdx.x) >> 5;   // 0..511
    int m  = lane & 15, kg = lane >> 4;

    const float* xrow = x + (size_t)(wave * 16 + m) * C_;
    const v16h*  wf   = (const v16h*)Wswz + lane;

    v8f acc[12];
#pragma unroll
    for (int ct = 0; ct < 12; ++ct) acc[ct] = (v8f){};

    for (int kc = 0; kc < 32; ++kc) {
        if (kc + 2 < 32)
            __builtin_prefetch(xrow + (kc + 2) * 32, 0, 0);
        // A fragment: 4 x b128 of f32 x, packed to f16.
        v16h a;
#pragma unroll
        for (int j = 0; j < 8; ++j) {
            int k0 = kc * 32 + a_kidx(j, kg);
            float2 f = *(const float2*)(xrow + k0);
            a[2 * j]     = (_Float16)f.x;
            a[2 * j + 1] = (_Float16)f.y;
        }
        // All 12 B fragments up front (distinct registers -> one load clause).
        v16h bfr[12];
#pragma unroll
        for (int ct = 0; ct < 12; ++ct)
            bfr[ct] = wf[(size_t)(kc * 12 + ct) * 32];
#pragma unroll
        for (int ct = 0; ct < 12; ++ct)
            acc[ct] = __builtin_amdgcn_wmma_f32_16x16x32_f16(
                          false, a, false, bfr[ct], (short)0, acc[ct],
                          false, false);
    }

#pragma unroll
    for (int ct = 0; ct < 12; ++ct) {
        _Float16* dst = (ct < 4) ? Q : (ct < 8 ? K : V);
        float     scl = (ct < 4) ? QSCALE : 1.0f;
        int       col = (ct & 3) * 16 + m;     // D-frag: N = lane&15
#pragma unroll
        for (int r = 0; r < 8; ++r) {
            int row = wave * 16 + r + 8 * kg;  // D-frag: M = r + 8*kg
            dst[(size_t)row * H_ + col] = (_Float16)(acc[ct][r] * scl);
        }
    }
}

// ---------------------------------------------------------------------------
// Kernel 3: swizzle V (f16 row-major) into B-fragment layout per
// (batch, 32-key chunk, h-tile). Fragment idx = ((b*64+s)*4+hc)*32+lane.
// ---------------------------------------------------------------------------
__global__ void __launch_bounds__(256)
swz_v(const _Float16* __restrict__ V, _Float16* __restrict__ Vswz) {
    int tid = blockIdx.x * blockDim.x + threadIdx.x;
    if (tid >= B_ * 64 * 4 * 32) return;
    int lane = tid & 31;
    int hc   = (tid >> 5) & 3;
    int s    = (tid >> 7) & 63;
    int b    = tid >> 13;
    int n = lane & 15, kg = lane >> 4;
    v16h o;
#pragma unroll
    for (int j = 0; j < 8; ++j) {
        int key = s * 32 + kg * 16 + 2 * j;
        o[2 * j]     = V[(size_t)(b * T_ + key)     * H_ + hc * 16 + n];
        o[2 * j + 1] = V[(size_t)(b * T_ + key + 1) * H_ + hc * 16 + n];
    }
    ((v16h*)Vswz)[((size_t)(b * 64 + s) * 4 + hc) * 32 + lane] = o;
}

// ---------------------------------------------------------------------------
// Kernel 4: causal flash attention. One wave per 16-query tile
// (512 waves, 4 waves/block). Per 32-key chunk:
//   S  : 4x WMMA (two 16-key tiles x two 32-wide h chunks)
//   softmax: fp32 online (m,l), 16-lane shfl_xor reductions
//   P->A-frag via 1KB per-wave LDS bounce
//   PV : 4x WMMA from V fragments prefetched one chunk ahead via
//        global_load_async_to_lds_b128 (double-buffered, s_wait_asynccnt).
// ---------------------------------------------------------------------------
__global__ void __launch_bounds__(128, 1)
attn(const _Float16* __restrict__ Q, const _Float16* __restrict__ K,
     const _Float16* __restrict__ Vswz, float* __restrict__ out) {
    __shared__ _Float16 plds[4][16 * 32];
    __shared__ __align__(16) _Float16 vstage[4][2][4 * 32 * 16]; // 32 KB

    int lane = threadIdx.x & 31;
    int wv   = threadIdx.x >> 5;
    int wave = blockIdx.x * 4 + wv;
    int b    = wave >> 7;       // 128 query tiles per batch
    int qt   = wave & 127;
    int m = lane & 15, kg = lane >> 4;
    _Float16* pl = plds[wv];

    // Q A-fragments (h chunks 0..31 and 32..63); rows are contiguous f16.
    const _Float16* qrow = Q + (size_t)(b * T_ + qt * 16 + m) * H_;
    v16h qa0, qa1;
#pragma unroll
    for (int j = 0; j < 8; ++j) {
        int k0 = a_kidx(j, kg);
        qa0[2 * j]     = qrow[k0];
        qa0[2 * j + 1] = qrow[k0 + 1];
        qa1[2 * j]     = qrow[32 + k0];
        qa1[2 * j + 1] = qrow[32 + k0 + 1];
    }

    v8f o0 = {}, o1 = {}, o2 = {}, o3 = {};
    float mrun[8], lrun[8];
#pragma unroll
    for (int r = 0; r < 8; ++r) { mrun[r] = -1e30f; lrun[r] = 0.0f; }

    const int nch = (qt >> 1) + 1;          // 32-key chunks needed (causal)

    // Prefetch chunk 0's four V fragments (8 async b128 ops -> ASYNCcnt=8).
    {
        const v16h* vb = (const v16h*)Vswz + ((size_t)(b * 64) * 4) * 32 + lane;
        _Float16* st = &vstage[wv][0][lane * 16];
        vfrag_async(vb +  0, st +  0 * 32 * 16);
        vfrag_async(vb + 32, st +  1 * 32 * 16);
        vfrag_async(vb + 64, st +  2 * 32 * 16);
        vfrag_async(vb + 96, st +  3 * 32 * 16);
    }

    for (int s = 0; s < nch; ++s) {
        // Prefetch next chunk's V fragments into the other buffer.
        if (s + 1 < nch) {
            const v16h* vb = (const v16h*)Vswz +
                             ((size_t)(b * 64 + s + 1) * 4) * 32 + lane;
            _Float16* st = &vstage[wv][(s + 1) & 1][lane * 16];
            vfrag_async(vb +  0, st +  0 * 32 * 16);
            vfrag_async(vb + 32, st +  1 * 32 * 16);
            vfrag_async(vb + 64, st +  2 * 32 * 16);
            vfrag_async(vb + 96, st +  3 * 32 * 16);
        }

        // K^T B-fragments: lane's N = key-in-tile (= m), K dim = h.
        const _Float16* kr0 = K + (size_t)(b * T_ + (2 * s)     * 16 + m) * H_;
        const _Float16* kr1 = K + (size_t)(b * T_ + (2 * s + 1) * 16 + m) * H_;
        v16h kb00, kb01, kb10, kb11;
#pragma unroll
        for (int j = 0; j < 8; ++j) {
            int kk = kg * 16 + 2 * j;
            kb00[2 * j] = kr0[kk];      kb00[2 * j + 1] = kr0[kk + 1];
            kb01[2 * j] = kr0[32 + kk]; kb01[2 * j + 1] = kr0[32 + kk + 1];
            kb10[2 * j] = kr1[kk];      kb10[2 * j + 1] = kr1[kk + 1];
            kb11[2 * j] = kr1[32 + kk]; kb11[2 * j + 1] = kr1[32 + kk + 1];
        }
        v8f s0 = {}, s1 = {};
        s0 = __builtin_amdgcn_wmma_f32_16x16x32_f16(false, qa0, false, kb00,
                                                    (short)0, s0, false, false);
        s0 = __builtin_amdgcn_wmma_f32_16x16x32_f16(false, qa1, false, kb01,
                                                    (short)0, s0, false, false);
        s1 = __builtin_amdgcn_wmma_f32_16x16x32_f16(false, qa0, false, kb10,
                                                    (short)0, s1, false, false);
        s1 = __builtin_amdgcn_wmma_f32_16x16x32_f16(false, qa1, false, kb11,
                                                    (short)0, s1, false, false);

        if (s == nch - 1) {                 // causal mask on diagonal chunk
#pragma unroll
            for (int r = 0; r < 8; ++r) {
                int q_abs = qt * 16 + r + 8 * kg;
                int k_abs = s * 32 + m;     // D-frag N = lane&15
                if (k_abs      > q_abs) s0[r] = -1e30f;
                if (k_abs + 16 > q_abs) s1[r] = -1e30f;
            }
        }

        // Online softmax in fp32 (rows live on 16-lane groups; kg bit fixed).
#pragma unroll
        for (int r = 0; r < 8; ++r) {
            float t0 = s0[r], t1 = s1[r];
            float mx = fmaxf(t0, t1);
            mx = fmaxf(mx, __shfl_xor(mx, 1, 32));
            mx = fmaxf(mx, __shfl_xor(mx, 2, 32));
            mx = fmaxf(mx, __shfl_xor(mx, 4, 32));
            mx = fmaxf(mx, __shfl_xor(mx, 8, 32));
            float mnew = fmaxf(mrun[r], mx);
            float al   = __expf(mrun[r] - mnew);
            float p0   = __expf(t0 - mnew);
            float p1   = __expf(t1 - mnew);
            float rs   = p0 + p1;
            rs += __shfl_xor(rs, 1, 32);
            rs += __shfl_xor(rs, 2, 32);
            rs += __shfl_xor(rs, 4, 32);
            rs += __shfl_xor(rs, 8, 32);
            lrun[r] = lrun[r] * al + rs;
            mrun[r] = mnew;
            o0[r] *= al; o1[r] *= al; o2[r] *= al; o3[r] *= al;
            int rowm = r + 8 * kg;
            pl[rowm * 32 + m]      = (_Float16)p0;
            pl[rowm * 32 + 16 + m] = (_Float16)p1;
        }

        // Reshape P (D-frag) -> A-frag (16 queries x 32 keys) through LDS.
        v16h pa;
#pragma unroll
        for (int j = 0; j < 8; ++j) {
            int k0 = a_kidx(j, kg);
            pa[2 * j]     = pl[m * 32 + k0];
            pa[2 * j + 1] = pl[m * 32 + k0 + 1];
        }

        // Wait for this chunk's async V stage (8 newer ops may remain).
        if (s + 1 < nch) asm volatile("s_wait_asynccnt 0x8" ::: "memory");
        else             asm volatile("s_wait_asynccnt 0x0" ::: "memory");

        const v16h* vs = (const v16h*)&vstage[wv][s & 1][0] + lane;
        o0 = __builtin_amdgcn_wmma_f32_16x16x32_f16(false, pa, false, vs[0],
                                                    (short)0, o0, false, false);
        o1 = __builtin_amdgcn_wmma_f32_16x16x32_f16(false, pa, false, vs[32],
                                                    (short)0, o1, false, false);
        o2 = __builtin_amdgcn_wmma_f32_16x16x32_f16(false, pa, false, vs[64],
                                                    (short)0, o2, false, false);
        o3 = __builtin_amdgcn_wmma_f32_16x16x32_f16(false, pa, false, vs[96],
                                                    (short)0, o3, false, false);
    }

    // Normalize by l and write fp32 output.
#pragma unroll
    for (int r = 0; r < 8; ++r) {
        float inv = 1.0f / lrun[r];
        size_t row = (size_t)(b * T_ + qt * 16 + r + 8 * kg) * H_;
        out[row +  0 + m] = o0[r] * inv;
        out[row + 16 + m] = o1[r] * inv;
        out[row + 32 + m] = o2[r] * inv;
        out[row + 48 + m] = o3[r] * inv;
    }
}

// ---------------------------------------------------------------------------
// Host launcher. Workspace layout (all 256B-aligned):
//   [0)        Wswz  : 12*32*32*16 f16 = 384 KB
//   [384K)     Q f16 : 8192*64          = 1 MB   (pre-scaled by 1/sqrt(C))
//   [+1M)      K f16 : 1 MB
//   [+2M)      V f16 : 1 MB
//   [+3M)      Vswz  : 1 MB
// ---------------------------------------------------------------------------
extern "C" void kernel_launch(void* const* d_in, const int* in_sizes, int n_in,
                              void* d_out, int out_size, void* d_ws,
                              size_t ws_size, hipStream_t stream) {
    (void)in_sizes; (void)n_in; (void)out_size; (void)ws_size;
    const float* x  = (const float*)d_in[0];
    const float* Wq = (const float*)d_in[1];
    const float* Wk = (const float*)d_in[2];
    const float* Wv = (const float*)d_in[3];
    float* out = (float*)d_out;

    char* ws = (char*)d_ws;
    const size_t OFF_Q = 393216;               // 384 KB
    const size_t OFF_K = OFF_Q + 1048576;
    const size_t OFF_V = OFF_K + 1048576;
    const size_t OFF_S = OFF_V + 1048576;
    _Float16* Wswz = (_Float16*)(ws);
    _Float16* Qh   = (_Float16*)(ws + OFF_Q);
    _Float16* Kh   = (_Float16*)(ws + OFF_K);
    _Float16* Vh   = (_Float16*)(ws + OFF_V);
    _Float16* Vswz = (_Float16*)(ws + OFF_S);

    swz_w   <<<48,  256, 0, stream>>>(Wq, Wk, Wv, Wswz);
    qkv_proj<<<64,  256, 0, stream>>>(x, Wswz, Qh, Kh, Vh);
    swz_v   <<<128, 256, 0, stream>>>(Vh, Vswz);
    attn    <<<128, 128, 0, stream>>>(Qh, Kh, Vswz, out);
}
